// DSSEncoder_17824114278579
// MI455X (gfx1250) — matrix operand order repaired and checked by reading, not compiled, not run on hardware
//
#include <hip/hip_runtime.h>
#include <math.h>

#define Bsz 512
#define Ssz 200
#define Ksz 16
#define Hsz 256
#define SPAD 224   // S padded to 7*32 for GEMM2 K-dim
#define MPAD 208   // S padded to 13*16 for M tiles
#define SZT  264   // half-stride for zt/zf rows (bank-conflict padding)
#define SATT 232   // half-stride for attnT rows

typedef __attribute__((ext_vector_type(16))) _Float16 v16h;
typedef __attribute__((ext_vector_type(8)))  _Float16 v8h;
typedef __attribute__((ext_vector_type(4)))  _Float16 v4h;
typedef __attribute__((ext_vector_type(8)))  float    v8f;

// ---- LDS layout (bytes) ----
#define ZT_OFF   0
#define ZF_OFF   (ZT_OFF + MPAD*SZT*2)      // keys_tilde f16 [208][264]
#define AT_OFF   (ZF_OFF + SPAD*SZT*2)      // z f16 [224][264]
#define SC_OFF   (AT_OFF + Ksz*SATT*2)      // attn^T f16 [16][232]
#define PIB_OFF  (SC_OFF + 1024)            // scores f32 [224]
#define RZ_OFF   (PIB_OFF + 1024)           // p_i f32 [224]
#define MU_OFF   (RZ_OFF + 1024)            // rz f32 [224]
#define IN_OFF   (MU_OFF + 1024)            // muz f32 [224]
#define Q_OFF    (IN_OFF + 1024)            // 1/||zn|| f32 [224]
#define WB_OFF   (Q_OFF + 1024)             // query f32 [256]
#define C12_OFF  (WB_OFF + 1024)            // w_bias f32 [256]
#define ENC_OFF  (C12_OFF + 256)            // c1/c2 f32 [32]
#define RED_OFF  (ENC_OFF + Ksz*Hsz*4)      // enc f32 [16][256]
#define LDS_BYTES (RED_OFF + 256)           // + reduce scratch
static_assert(LDS_BYTES <= 320*1024, "LDS budget");

// ---- batched cross-lane reductions: N independent chains, stage-outer ----
template<int N>
__device__ __forceinline__ void wred_sum_b(float* v) {
#pragma unroll
  for (int m = 16; m >= 1; m >>= 1) {
    float t[N];
#pragma unroll
    for (int i = 0; i < N; ++i) t[i] = __shfl_xor(v[i], m, 32);
#pragma unroll
    for (int i = 0; i < N; ++i) v[i] += t[i];
  }
}
template<int N>
__device__ __forceinline__ void hred_sum_b(float* v) {  // within 16-lane halves
#pragma unroll
  for (int m = 8; m >= 1; m >>= 1) {
    float t[N];
#pragma unroll
    for (int i = 0; i < N; ++i) t[i] = __shfl_xor(v[i], m, 32);
#pragma unroll
    for (int i = 0; i < N; ++i) v[i] += t[i];
  }
}
template<int N>
__device__ __forceinline__ void hred_max_b(float* v) {
#pragma unroll
  for (int m = 8; m >= 1; m >>= 1) {
    float t[N];
#pragma unroll
    for (int i = 0; i < N; ++i) t[i] = __shfl_xor(v[i], m, 32);
#pragma unroll
    for (int i = 0; i < N; ++i) v[i] = fmaxf(v[i], t[i]);
  }
}
__device__ __forceinline__ float wred_sum(float v) {
#pragma unroll
  for (int m = 16; m >= 1; m >>= 1) v += __shfl_xor(v, m, 32);
  return v;
}
__device__ __forceinline__ float wred_max(float v) {
#pragma unroll
  for (int m = 16; m >= 1; m >>= 1) v = fmaxf(v, __shfl_xor(v, m, 32));
  return v;
}
__device__ __forceinline__ void up4(float4 a, float4 b, float* o) {
  o[0]=a.x; o[1]=a.y; o[2]=a.z; o[3]=a.w; o[4]=b.x; o[5]=b.y; o[6]=b.z; o[7]=b.w;
}
__device__ __forceinline__ float4 pk4(const float* o) {
  float4 r; r.x=o[0]; r.y=o[1]; r.z=o[2]; r.w=o[3]; return r;
}

// ---- prep: pn = LN2(proto)/max(||.||,1e-6);
//      U[k][h] = ln1w[h]*pn[k][h] (f16, WMMA-B layout), c1[k]=sum U, c2[k]=dot(ln1b,pn_k)
__global__ __launch_bounds__(512) void prep_pn(const float* __restrict__ proto,
                                               const float* __restrict__ ln2w,
                                               const float* __restrict__ ln2b,
                                               const float* __restrict__ ln1w,
                                               const float* __restrict__ ln1b,
                                               _Float16* __restrict__ U,
                                               float* __restrict__ c12) {
  const int wid = threadIdx.x >> 5, lane = threadIdx.x & 31;  // wid = proto row 0..15
  const float4* row = (const float4*)(proto + (size_t)wid * Hsz);
  float xv[8]; up4(row[lane], row[32+lane], xv);
  float s = 0.f, ss = 0.f;
#pragma unroll
  for (int j = 0; j < 8; ++j) { s += xv[j]; ss += xv[j]*xv[j]; }
  float r2[2] = {s, ss}; wred_sum_b<2>(r2);
  float mu = r2[0] * (1.f/Hsz);
  float rs = rsqrtf(r2[1] * (1.f/Hsz) - mu*mu + 1e-12f);
  float wv[8], bv[8];
  up4(((const float4*)ln2w)[lane], ((const float4*)ln2w)[32+lane], wv);
  up4(((const float4*)ln2b)[lane], ((const float4*)ln2b)[32+lane], bv);
  float pv[8]; float n2 = 0.f;
#pragma unroll
  for (int j = 0; j < 8; ++j) { pv[j] = (xv[j]-mu)*rs*wv[j] + bv[j]; n2 += pv[j]*pv[j]; }
  n2 = wred_sum(n2);
  float inv = 1.0f / fmaxf(sqrtf(n2), 1e-6f);
  float l1w[8], l1b[8];
  up4(((const float4*)ln1w)[lane], ((const float4*)ln1w)[32+lane], l1w);
  up4(((const float4*)ln1b)[lane], ((const float4*)ln1b)[32+lane], l1b);
  float cc[2] = {0.f, 0.f};
#pragma unroll
  for (int j = 0; j < 8; ++j) {
    pv[j] *= inv;
    cc[0] += l1w[j]*pv[j];
    cc[1] += l1b[j]*pv[j];
  }
  wred_sum_b<2>(cc);
#pragma unroll
  for (int j = 0; j < 8; ++j) {
    int h = (j < 4) ? (4*lane + j) : (128 + 4*lane + (j-4));
    U[(size_t)wid*Hsz + h] = (_Float16)(l1w[j]*pv[j]);
  }
  if (lane == 0) { c12[wid] = cc[0]; c12[16+wid] = cc[1]; }
}

// ---- prep: Wt[n][h] = (f16) W[n][h]  (WMMA-B friendly layout)
__global__ __launch_bounds__(256) void prep_w(const float* __restrict__ W,
                                              _Float16* __restrict__ Wt) {
  int idx = blockIdx.x * 256 + threadIdx.x;   // idx = n*256 + h (coalesced read)
  int h = idx & 255, n = idx >> 8;
  Wt[(size_t)n * Hsz + h] = (_Float16)W[idx];
}

// ---- main fused kernel: one block per batch ----
__global__ __launch_bounds__(256) void dss_main(
    const float* __restrict__ z,
    const float* __restrict__ ln1w, const float* __restrict__ ln1b,
    const float* __restrict__ ln3w, const float* __restrict__ ln3b,
    const float* __restrict__ ln4w, const float* __restrict__ ln4b,
    const float* __restrict__ ln5w, const float* __restrict__ ln5b,
    const float* __restrict__ w_bias, const float* __restrict__ b_prime,
    const float* __restrict__ alphas,
    const float* __restrict__ betaI, const float* __restrict__ betaL,
    const int*   __restrict__ is_input,
    const _Float16* __restrict__ U, const float* __restrict__ c12g,
    const _Float16* __restrict__ Wt,
    float* __restrict__ outp) {
  extern __shared__ char smem[];
  _Float16* zt  = (_Float16*)(smem + ZT_OFF);
  _Float16* zf  = (_Float16*)(smem + ZF_OFF);
  _Float16* at  = (_Float16*)(smem + AT_OFF);
  float* sco  = (float*)(smem + SC_OFF);
  float* pib  = (float*)(smem + PIB_OFF);
  float* rzb  = (float*)(smem + RZ_OFF);
  float* muzb = (float*)(smem + MU_OFF);
  float* invb = (float*)(smem + IN_OFF);
  float* qb   = (float*)(smem + Q_OFF);
  float* wbs  = (float*)(smem + WB_OFF);
  float* c12l = (float*)(smem + C12_OFF);
  float* enc  = (float*)(smem + ENC_OFF);
  float* red  = (float*)(smem + RED_OFF);

  const int tid  = threadIdx.x;
  const int lane = tid & 31;
  const int wid  = tid >> 5;
  const int b    = blockIdx.x;
  const int hlf  = lane >> 4;
  const int col  = lane & 15;
  const float inv_sqrt_h = 0.0625f;   // 1/sqrt(256)

  // ================= phase 0: copies, zero-padding, query =================
  {
    wbs[tid] = w_bias[tid];
    if (tid < 32) c12l[tid] = c12g[tid];
    if (tid < 24) { rzb[200+tid] = 0.f; muzb[200+tid] = 0.f; invb[200+tid] = 0.f; }
    unsigned* pz1 = (unsigned*)(zt + 200*SZT);
    for (int i = tid; i < 8*SZT/2; i += 256) pz1[i] = 0u;
    unsigned* pz2 = (unsigned*)(zf + 200*SZT);
    for (int i = tid; i < 24*SZT/2; i += 256) pz2[i] = 0u;
    unsigned* pz3 = (unsigned*)at;
    for (int i = tid; i < Ksz*SATT/2; i += 256) pz3[i] = 0u;
    if (wid == 0) {  // query = LN4(b_prime + alphas[-1] + z[:, -1, :])
      const float4* zr = (const float4*)(z + ((size_t)b*Ssz + (Ssz-1))*Hsz);
      const float4* ar = (const float4*)(alphas + (size_t)(Ssz-1)*Hsz);
      const float4* br = (const float4*)b_prime;
      float zv[8], av[8], pv[8];
      up4(zr[lane], zr[32+lane], zv); up4(ar[lane], ar[32+lane], av);
      up4(br[lane], br[32+lane], pv);
      float xv[8], sx = 0.f, sxx = 0.f;
#pragma unroll
      for (int j = 0; j < 8; ++j) { xv[j] = zv[j]+av[j]+pv[j]; sx += xv[j]; sxx += xv[j]*xv[j]; }
      float r2[2] = {sx, sxx}; wred_sum_b<2>(r2);
      float mu = r2[0]*(1.f/Hsz), rs = rsqrtf(r2[1]*(1.f/Hsz) - mu*mu + 1e-12f);
      float wv[8], bv[8];
      up4(((const float4*)ln4w)[lane], ((const float4*)ln4w)[32+lane], wv);
      up4(((const float4*)ln4b)[lane], ((const float4*)ln4b)[32+lane], bv);
      float ov[8];
#pragma unroll
      for (int j = 0; j < 8; ++j) ov[j] = (xv[j]-mu)*rs*wv[j] + bv[j];
      ((float4*)qb)[lane] = pk4(ov); ((float4*)qb)[32+lane] = pk4(ov+4);
    }
  }
  __syncthreads();

  // ====== phase B: per-row LN1/LN3 stats, stage z & keys_tilde (f16) ======
  {
    float l1w[8], l1b[8], l3w[8], l3b[8], qv[8];
    up4(((const float4*)ln1w)[lane], ((const float4*)ln1w)[32+lane], l1w);
    up4(((const float4*)ln1b)[lane], ((const float4*)ln1b)[32+lane], l1b);
    up4(((const float4*)ln3w)[lane], ((const float4*)ln3w)[32+lane], l3w);
    up4(((const float4*)ln3b)[lane], ((const float4*)ln3b)[32+lane], l3b);
    up4(((const float4*)qb)[lane], ((const float4*)qb)[32+lane], qv);
    for (int it = 0; it < 25; ++it) {
      int s = wid + 8*it;                              // 0..199
      const float4* zr = (const float4*)(z + ((size_t)b*Ssz + s)*Hsz);
      const float4* ar = (const float4*)(alphas + (size_t)s*Hsz);
      float zv[8], av[8];
      up4(zr[lane], zr[32+lane], zv); up4(ar[lane], ar[32+lane], av);
      float xv[8], r4[4] = {0.f, 0.f, 0.f, 0.f};
#pragma unroll
      for (int j = 0; j < 8; ++j) {
        xv[j] = zv[j] + av[j];
        r4[0] += zv[j]; r4[1] += zv[j]*zv[j]; r4[2] += xv[j]; r4[3] += xv[j]*xv[j];
      }
      wred_sum_b<4>(r4);
      float muz = r4[0]*(1.f/Hsz), rz = rsqrtf(r4[1]*(1.f/Hsz)-muz*muz + 1e-12f);
      float mux = r4[2]*(1.f/Hsz), rx = rsqrtf(r4[3]*(1.f/Hsz)-mux*mux + 1e-12f);
      float kt[8], r2[2] = {0.f, 0.f};
#pragma unroll
      for (int j = 0; j < 8; ++j) {
        float zn = (zv[j]-muz)*rz*l1w[j] + l1b[j]; r2[0] += zn*zn;
        kt[j] = (xv[j]-mux)*rx*l3w[j] + l3b[j];    r2[1] += kt[j]*qv[j];
      }
      wred_sum_b<2>(r2);
      if (lane == 0) {
        sco[s]  = r2[1];                              // keys_tilde . query part
        rzb[s]  = rz;
        muzb[s] = muz;
        invb[s] = 1.0f / fmaxf(sqrtf(r2[0]), 1e-6f);  // 1/||zn||
      }
      v4h h0, h1, g0, g1;
#pragma unroll
      for (int j = 0; j < 4; ++j) {
        h0[j] = (_Float16)kt[j]; h1[j] = (_Float16)kt[j+4];
        g0[j] = (_Float16)zv[j]; g1[j] = (_Float16)zv[j+4];
      }
      *(v4h*)(zt + s*SZT + 4*lane)       = h0;
      *(v4h*)(zt + s*SZT + 128 + 4*lane) = h1;
      *(v4h*)(zf + s*SZT + 4*lane)       = g0;
      *(v4h*)(zf + s*SZT + 128 + 4*lane) = g1;
    }
  }
  __syncthreads();

  // ===== GEMM1: relu(keys_tilde @ W^T + bias) . query, fused into scores =====
  for (int t = 0; t < 2; ++t) {
    const int n0 = (wid*2 + t) * 16;
    v16h Bs[8];
#pragma unroll
    for (int kk = 0; kk < 8; ++kk) {                     // B: lane%16 = N col, 16 K halfs (table order)
      const _Float16* bp = Wt + (size_t)(n0 + col)*Hsz + kk*32 + hlf*8;
      v8h blo = *(const v8h*)bp;
      v8h bhi = *(const v8h*)(bp + 16);
      Bs[kk] = __builtin_shufflevector(blo, bhi, 0,1,2,3,4,5,6,7,8,9,10,11,12,13,14,15);
    }
    const float wb_c = wbs[n0 + col];
    const float q_c  = qb[n0 + col];
    for (int mt = 0; mt < 13; ++mt) {
      const int m0 = mt*16;
      v8f acc = {0.f,0.f,0.f,0.f,0.f,0.f,0.f,0.f};
      const _Float16* arow = zt + (m0 + col)*SZT;
#pragma unroll
      for (int kk = 0; kk < 8; ++kk) {
        const int k0 = kk*32;
        v8h alo = *(const v8h*)(arow + k0 + hlf*8);
        v8h ahi = *(const v8h*)(arow + k0 + 16 + hlf*8);
        v16h a = __builtin_shufflevector(alo, ahi, 0,1,2,3,4,5,6,7,8,9,10,11,12,13,14,15);
        acc = __builtin_amdgcn_wmma_f32_16x16x32_f16(false, a, false, Bs[kk],
                                                     (short)0, acc, false, false);
      }
      float v[8];
#pragma unroll
      for (int r = 0; r < 8; ++r) v[r] = fmaxf(acc[r] + wb_c, 0.0f) * q_c;
      hred_sum_b<8>(v);                                  // 4 batched stages
      if (col == 0) {
#pragma unroll
        for (int r = 0; r < 8; ++r) atomicAdd(&sco[m0 + r + 8*hlf], v[r]);
      }
    }
  }
  __syncthreads();

  // ================= softmax over S -> p_i =================
  {
    float val = (tid < Ssz) ? sco[tid]*inv_sqrt_h : -1e30f;
    float wm = wred_max(val);
    if (lane == 0) red[wid] = wm;
    __syncthreads();
    float gm = red[0];
#pragma unroll
    for (int i = 1; i < 8; ++i) gm = fmaxf(gm, red[i]);
    float e = (tid < Ssz) ? __expf(val - gm) : 0.0f;
    float ws = wred_sum(e);
    if (lane == 0) red[8+wid] = ws;
    __syncthreads();
    float gs = red[8];
#pragma unroll
    for (int i = 1; i < 8; ++i) gs += red[8+i];
    if (tid < SPAD) pib[tid] = (tid < Ssz) ? e/gs : 0.0f;
  }
  __syncthreads();

  // ===== GEMM-C: cos = (rz*(z@U - muz*c1) + c2)/||zn||; softmax_K; *p_i -> attn^T =====
  {
    v16h Us[8];
#pragma unroll
    for (int kk = 0; kk < 8; ++kk) {                     // B strip: U [16][256]
      const _Float16* up = U + (size_t)col*Hsz + kk*32 + hlf*8;
      v8h blo = *(const v8h*)up;
      v8h bhi = *(const v8h*)(up + 16);
      Us[kk] = __builtin_shufflevector(blo, bhi, 0,1,2,3,4,5,6,7,8,9,10,11,12,13,14,15);
    }
    const float c1c = c12l[col], c2c = c12l[16+col];
    for (int mt = wid; mt < 13; mt += 8) {
      const int m0 = mt*16;
      v8f acc = {0.f,0.f,0.f,0.f,0.f,0.f,0.f,0.f};
      const _Float16* arow = zf + (m0 + col)*SZT;
#pragma unroll
      for (int kk = 0; kk < 8; ++kk) {
        const int k0 = kk*32;
        v8h alo = *(const v8h*)(arow + k0 + hlf*8);
        v8h ahi = *(const v8h*)(arow + k0 + 16 + hlf*8);
        v16h a = __builtin_shufflevector(alo, ahi, 0,1,2,3,4,5,6,7,8,9,10,11,12,13,14,15);
        acc = __builtin_amdgcn_wmma_f32_16x16x32_f16(false, a, false, Us[kk],
                                                     (short)0, acc, false, false);
      }
      float lg[8], mx[8];
#pragma unroll
      for (int r = 0; r < 8; ++r) {
        const int si = m0 + r + 8*hlf;
        lg[r] = (rzb[si]*(acc[r] - muzb[si]*c1c) + c2c) * invb[si] * inv_sqrt_h;
        mx[r] = lg[r];
      }
      hred_max_b<8>(mx);                                 // max over 16 protos
      float ex[8], sm[8];
#pragma unroll
      for (int r = 0; r < 8; ++r) { ex[r] = __expf(lg[r]-mx[r]); sm[r] = ex[r]; }
      hred_sum_b<8>(sm);
#pragma unroll
      for (int r = 0; r < 8; ++r) {
        const int si = m0 + r + 8*hlf;
        float val = (si < Ssz) ? (ex[r]/sm[r]) * pib[si] : 0.0f;
        at[col*SATT + si] = (_Float16)val;
      }
    }
  }
  __syncthreads();

  // ===== GEMM2: enc[k][h] = beta + sum_s attn[s][k] * z[s][h]; M=16, K=224 =====
  {
    const float* beta = (is_input[0] != 0) ? betaI : betaL;
    for (int t = 0; t < 2; ++t) {
      const int n0 = (wid*2 + t) * 16;
      v8f acc = {0.f,0.f,0.f,0.f,0.f,0.f,0.f,0.f};
      const _Float16* arow = at + col*SATT;              // row = proto index
#pragma unroll
      for (int kk = 0; kk < 7; ++kk) {
        const int s0 = kk*32;
        v8h alo = *(const v8h*)(arow + s0 + hlf*8);
        v8h ahi = *(const v8h*)(arow + s0 + 16 + hlf*8);
        v16h a = __builtin_shufflevector(alo, ahi, 0,1,2,3,4,5,6,7,8,9,10,11,12,13,14,15);
        const _Float16* brow = zf + (s0 + hlf*8)*SZT + n0 + col;
        v16h bm;
#pragma unroll
        for (int j = 0; j < 8; ++j) { bm[j] = brow[j*SZT]; bm[8+j] = brow[(16+j)*SZT]; }
        acc = __builtin_amdgcn_wmma_f32_16x16x32_f16(false, a, false, bm,
                                                     (short)0, acc, false, false);
      }
#pragma unroll
      for (int r = 0; r < 8; ++r) {
        const int p = r + 8*hlf;
        const int n = n0 + col;
        enc[p*Hsz + n] = acc[r] + beta[(size_t)p*Hsz + n];
      }
    }
  }
  __syncthreads();

  // ================= LN5(enc) -> out =================
  for (int t = 0; t < 2; ++t) {
    const int p = wid*2 + t;
    const float4* er = (const float4*)(enc + p*Hsz);
    float ev[8]; up4(er[lane], er[32+lane], ev);
    float r2[2] = {0.f, 0.f};
#pragma unroll
    for (int j = 0; j < 8; ++j) { r2[0] += ev[j]; r2[1] += ev[j]*ev[j]; }
    wred_sum_b<2>(r2);
    float mu = r2[0]*(1.f/Hsz), rs = rsqrtf(r2[1]*(1.f/Hsz)-mu*mu + 1e-12f);
    float wv[8], bv[8];
    up4(((const float4*)ln5w)[lane], ((const float4*)ln5w)[32+lane], wv);
    up4(((const float4*)ln5b)[lane], ((const float4*)ln5b)[32+lane], bv);
    float ov[8];
#pragma unroll
    for (int j = 0; j < 8; ++j) ov[j] = (ev[j]-mu)*rs*wv[j] + bv[j];
    float4* orow = (float4*)(outp + ((size_t)b*Ksz + p)*Hsz);
    orow[lane] = pk4(ov); orow[32+lane] = pk4(ov+4);
  }
}

extern "C" void kernel_launch(void* const* d_in, const int* in_sizes, int n_in,
                              void* d_out, int out_size, void* d_ws, size_t ws_size,
                              hipStream_t stream) {
  const float* z      = (const float*)d_in[0];
  const float* proto  = (const float*)d_in[1];
  const float* ln1w   = (const float*)d_in[2];
  const float* ln1b   = (const float*)d_in[3];
  const float* ln2w   = (const float*)d_in[4];
  const float* ln2b   = (const float*)d_in[5];
  const float* ln3w   = (const float*)d_in[6];
  const float* ln3b   = (const float*)d_in[7];
  const float* ln4w   = (const float*)d_in[8];
  const float* ln4b   = (const float*)d_in[9];
  const float* ln5w   = (const float*)d_in[10];
  const float* ln5b   = (const float*)d_in[11];
  const float* Wmat   = (const float*)d_in[12];
  const float* wbias  = (const float*)d_in[13];
  const float* bprime = (const float*)d_in[14];
  const float* alphas = (const float*)d_in[15];
  const float* betaI  = (const float*)d_in[16];
  const float* betaL  = (const float*)d_in[17];
  const int*   isin   = (const int*)d_in[18];
  float* outp = (float*)d_out;

  _Float16* U    = (_Float16*)d_ws;                           // 8 KB [16][256]
  float*    c12  = (float*)((char*)d_ws + Hsz*Ksz*2);         // 128 B
  _Float16* Wt   = (_Float16*)((char*)d_ws + 16384);          // 128 KB

  prep_pn<<<1, 512, 0, stream>>>(proto, ln2w, ln2b, ln1w, ln1b, U, c12);
  prep_w<<<Hsz, 256, 0, stream>>>(Wmat, Wt);
  hipFuncSetAttribute((const void*)dss_main,
                      hipFuncAttributeMaxDynamicSharedMemorySize, LDS_BYTES);
  dss_main<<<Bsz, 256, LDS_BYTES, stream>>>(
      z, ln1w, ln1b, ln3w, ln3b, ln4w, ln4b, ln5w, ln5b,
      wbias, bprime, alphas, betaI, betaL, isin, U, c12, Wt, outp);
}
